// MultiHeadAttention_78847009620030
// MI455X (gfx1250) — matrix operand order repaired and checked by reading, not compiled, not run on hardware
//
#include <hip/hip_runtime.h>
#include <hip/hip_bf16.h>

typedef __attribute__((ext_vector_type(16))) _Float16 v16h;
typedef __attribute__((ext_vector_type(8)))  _Float16 v8h;
typedef __attribute__((ext_vector_type(8)))  float    v8f;

#define NH   8
#define DH   64
#define BB   4
#define KK   2048
#define DD   512
#define MM   (BB*KK)   /* 8192 */
#define BAND 128

// ---------------------------------------------------------------- convert f32 -> f16
__global__ void cvt_f32_f16(const float* __restrict__ src, _Float16* __restrict__ dst, int n) {
    int i = blockIdx.x * blockDim.x + threadIdx.x;
    int stride = gridDim.x * blockDim.x;
    for (; i < n; i += stride) dst[i] = (_Float16)src[i];
}

// ---------------------------------------------------------------- Y = X @ W^T + b
// 32x64 tile per wave: 8 WMMAs per k-step against 8 fragment loads (4 A + 4 B),
// with each B fragment reused by two row-tiles in-register.
// mode 0: store f16 Y[m*DD+n]   (Q, K)
// mode 1: store f16 Y^T[n*MM+m] (V transposed for attention B-fragments)
// mode 2: store f32 Y[m*DD+n]   (final output projection)
__global__ __launch_bounds__(256) void gemm_xwt(const _Float16* __restrict__ X,
                                                const _Float16* __restrict__ W,
                                                const float* __restrict__ bias,
                                                void* __restrict__ out, int mode) {
    const int lane = threadIdx.x & 31;
    const int wv   = threadIdx.x >> 5;
    const int lx   = lane & 15;
    const int hi   = lane >> 4;
    const int wgid = blockIdx.x * 8 + wv;         // 2048 waves
    const int strip = wgid & 7;                   // 8 strips of 64 cols
    const int m0    = (wgid >> 3) * 32;           // 256 row tiles of 32
    const int n0    = strip * 64;

    union AF { v16h v; v8h h[2]; };
    const _Float16* arow0 = X + (m0 + lx) * DD + 16 * hi;       // rows m0..m0+15
    const _Float16* arow1 = arow0 + 16 * DD;                    // rows m0+16..m0+31
    const _Float16* wbase = W + (n0 + lx) * DD + 16 * hi;       // B-frag base (tile t: +t*16*DD)

    v8f acc[2][4] = {};
    for (int k0 = 0; k0 < DD; k0 += 32) {         // 16 k-steps
        AF a0, a1;
        a0.h[0] = *(const v8h*)(arow0 + k0);
        a0.h[1] = *(const v8h*)(arow0 + k0 + 8);
        a1.h[0] = *(const v8h*)(arow1 + k0);
        a1.h[1] = *(const v8h*)(arow1 + k0 + 8);
#pragma unroll
        for (int t = 0; t < 4; ++t) {
            v16h bf = *(const v16h*)(wbase + t * 16 * DD + k0);
            acc[0][t] = __builtin_amdgcn_wmma_f32_16x16x32_f16(
                false, a0.v, false, bf, (short)0, acc[0][t], false, false);
            acc[1][t] = __builtin_amdgcn_wmma_f32_16x16x32_f16(
                false, a1.v, false, bf, (short)0, acc[1][t], false, false);
        }
    }

#pragma unroll
    for (int u = 0; u < 2; ++u)
#pragma unroll
        for (int t = 0; t < 4; ++t) {
            const int n = n0 + t * 16 + lx;
            const float bv = bias[n];
#pragma unroll
            for (int r = 0; r < 8; ++r) {
                const int m = m0 + u * 16 + r + 8 * hi;
                const float y = acc[u][t][r] + bv;
                if (mode == 0)      ((_Float16*)out)[m * DD + n] = (_Float16)y;
                else if (mode == 1) ((_Float16*)out)[n * MM + m] = (_Float16)y;
                else                ((float*)out)[m * DD + n] = y;
            }
        }
}

// ---------------------------------------------------------------- banded flash attention
// One wave per (h, b, 16-query tile). S^T = K_tile x Q^T so softmax stats are
// lane-indexed and P^T C-layout registers are directly the P A-fragment.
__global__ __launch_bounds__(256) void attn_band(const _Float16* __restrict__ Q,
                                                 const _Float16* __restrict__ Km,
                                                 const _Float16* __restrict__ Vt,
                                                 _Float16* __restrict__ Oh) {
    const int lane = threadIdx.x & 31;
    const int wv   = threadIdx.x >> 5;
    const int lx   = lane & 15;
    const int hi   = lane >> 4;
    const int wgid = blockIdx.x * 8 + wv;     // 4096 waves = 8h * 4b * 128 tiles
    const int h   = wgid >> 9;
    const int b   = (wgid >> 7) & 3;
    const int it  = wgid & 127;
    const int i0  = it * 16;
    const float rscale = 0.022097086912079612f;   // 1/sqrt(2048) (ref scales by sqrt(seq_len))

    // Q as B-fragment: lane c holds Q[i0 + c%16][d0 + 16*hi .. +16]
    v16h qf[2];
    {
        const _Float16* qrow = Q + (b * KK + i0 + lx) * DD + h * DH;
        qf[0] = *(const v16h*)(qrow + 16 * hi);
        qf[1] = *(const v16h*)(qrow + 32 + 16 * hi);
    }

    v8f   o[4] = {};
    float m_run = -__builtin_inff();
    float l_run = 0.f;
    int jt_lo = it - 8; if (jt_lo < 0) jt_lo = 0;
    int jt_hi = it + 8; if (jt_hi > 127) jt_hi = 127;
    const int iq = i0 + lx;                   // this lane's query row

    for (int jt = jt_lo; jt <= jt_hi; jt += 2) {
        // V fragments for this 32-wide key slab (independent of S / softmax):
        // issue early so they overlap the S WMMAs and the exp chain.
        const int jc0 = jt * 16 + 8 * hi;
        int jc1 = jc0 + 16; if (jc1 > KK - 8) jc1 = KK - 8;  // clamped cols have P==0
        v16h vf[4];
#pragma unroll
        for (int t = 0; t < 4; ++t) {
            union { v16h v; v8h p[2]; } vu;
            vu.p[0] = *(const v8h*)(Vt + (h * DH + t * 16 + lx) * MM + b * KK + jc0);
            vu.p[1] = *(const v8h*)(Vt + (h * DH + t * 16 + lx) * MM + b * KK + jc1);
            vf[t] = vu.v;
        }

        // prefetch next slab's K rows (gfx1250 global_prefetch_b8)
        if (jt + 2 <= jt_hi) {
            const int jp = (jt + 2) * 16 + lx;
            __builtin_prefetch((const void*)(Km + (b * KK + jp) * DD + h * DH), 0, 1);
        }

        v8f s[2];
#pragma unroll
        for (int t = 0; t < 2; ++t) {
            const int jtt = jt + t;
            int jr = jtt * 16 + lx; if (jr > KK - 1) jr = KK - 1;   // clamped edge load
            const _Float16* krow = Km + (b * KK + jr) * DD + h * DH;
            v8f st = {};
#pragma unroll
            for (int kk = 0; kk < 2; ++kk) {
                union { v16h v; v8h p[2]; } af;    // K rows as A-fragment
                af.p[0] = *(const v8h*)(krow + kk * 32 + 16 * hi);
                af.p[1] = *(const v8h*)(krow + kk * 32 + 16 * hi + 8);
                st = __builtin_amdgcn_wmma_f32_16x16x32_f16(
                    false, af.v, false, qf[kk], (short)0, st, false, false);
            }
            // scale + band mask (element r is key j = jtt*16 + r + 8*hi)
#pragma unroll
            for (int r = 0; r < 8; ++r) {
                const int j = jtt * 16 + r + 8 * hi;
                int d = iq - j; if (d < 0) d = -d;
                const bool masked = (jtt > jt_hi) || (j >= KK) || (d > BAND);
                st[r] = masked ? -__builtin_inff() : st[r] * rscale;
            }
            s[t] = st;
        }
        // online softmax; stats per lane (query i = i0 + lx)
        float tmax = -__builtin_inff();
#pragma unroll
        for (int t = 0; t < 2; ++t)
#pragma unroll
            for (int r = 0; r < 8; ++r) tmax = fmaxf(tmax, s[t][r]);
        tmax = fmaxf(tmax, __shfl_xor(tmax, 16, 32));
        const float m_new = fmaxf(m_run, tmax);
        const float scl   = __expf(m_run - m_new);   // 0 on first iter (m_run = -inf)
        float tsum = 0.f;
        v16h pf;                                     // P A-fragment, built in-lane
#pragma unroll
        for (int t = 0; t < 2; ++t)
#pragma unroll
            for (int r = 0; r < 8; ++r) {
                const float p = __expf(s[t][r] - m_new);
                tsum += p;
                pf[t * 8 + r] = (_Float16)p;
            }
        tsum += __shfl_xor(tsum, 16, 32);
        l_run = l_run * scl + tsum;
        m_run = m_new;
        // rescale O: accumulator row r+8*hi needs stats of lane r+8*hi
#pragma unroll
        for (int r = 0; r < 8; ++r) {
            const float sr = __shfl(scl, r + 8 * hi, 32);
#pragma unroll
            for (int t = 0; t < 4; ++t) o[t][r] *= sr;
        }
        // O += P @ V  (V fragments already in registers)
#pragma unroll
        for (int t = 0; t < 4; ++t)
            o[t] = __builtin_amdgcn_wmma_f32_16x16x32_f16(
                false, pf, false, vf[t], (short)0, o[t], false, false);
    }
    // normalize and store heads as f16 [b][i][h*64+dv]
#pragma unroll
    for (int r = 0; r < 8; ++r) {
        const float lr = __shfl(l_run, r + 8 * hi, 32);
        const float inv = 1.0f / lr;
#pragma unroll
        for (int t = 0; t < 4; ++t)
            Oh[(b * KK + i0 + r + 8 * hi) * DD + h * DH + t * 16 + lx] =
                (_Float16)(o[t][r] * inv);
    }
}

// ---------------------------------------------------------------- launcher
extern "C" void kernel_launch(void* const* d_in, const int* in_sizes, int n_in,
                              void* d_out, int out_size, void* d_ws, size_t ws_size,
                              hipStream_t stream) {
    (void)in_sizes; (void)n_in; (void)out_size; (void)ws_size;
    const float* query = (const float*)d_in[0];
    const float* key   = (const float*)d_in[1];
    const float* value = (const float*)d_in[2];
    const float* Wq = (const float*)d_in[3];  const float* bq = (const float*)d_in[4];
    const float* Wk = (const float*)d_in[5];  const float* bk = (const float*)d_in[6];
    const float* Wv = (const float*)d_in[7];  const float* bv = (const float*)d_in[8];
    const float* Wo = (const float*)d_in[9];  const float* bo = (const float*)d_in[10];

    char* ws = (char*)d_ws;
    const size_t SZ_X = (size_t)MM * DD * sizeof(_Float16);   // 8 MB
    const size_t SZ_W = (size_t)DD * DD * sizeof(_Float16);   // 512 KB
    _Float16* q16  = (_Float16*)(ws);
    _Float16* k16  = (_Float16*)(ws + SZ_X);
    _Float16* v16  = (_Float16*)(ws + 2 * SZ_X);
    _Float16* wq16 = (_Float16*)(ws + 3 * SZ_X);
    _Float16* wk16 = (_Float16*)(ws + 3 * SZ_X + SZ_W);
    _Float16* wv16 = (_Float16*)(ws + 3 * SZ_X + 2 * SZ_W);
    _Float16* wo16 = (_Float16*)(ws + 3 * SZ_X + 3 * SZ_W);
    _Float16* Qh   = (_Float16*)(ws + 3 * SZ_X + 4 * SZ_W);
    _Float16* Kh   = (_Float16*)(ws + 4 * SZ_X + 4 * SZ_W);
    _Float16* Vt   = (_Float16*)(ws + 5 * SZ_X + 4 * SZ_W);
    _Float16* Ah   = (_Float16*)(ws + 6 * SZ_X + 4 * SZ_W);

    const int NX = MM * DD, NW = DD * DD;
    cvt_f32_f16<<<1024, 256, 0, stream>>>(query, q16, NX);
    cvt_f32_f16<<<1024, 256, 0, stream>>>(key,   k16, NX);
    cvt_f32_f16<<<1024, 256, 0, stream>>>(value, v16, NX);
    cvt_f32_f16<<<256, 256, 0, stream>>>(Wq, wq16, NW);
    cvt_f32_f16<<<256, 256, 0, stream>>>(Wk, wk16, NW);
    cvt_f32_f16<<<256, 256, 0, stream>>>(Wv, wv16, NW);
    cvt_f32_f16<<<256, 256, 0, stream>>>(Wo, wo16, NW);

    // 2048 waves / 8 per block = 256 blocks per GEMM (32-row tiles)
    gemm_xwt<<<256, 256, 0, stream>>>(q16, wq16, bq, (void*)Qh, 0);
    gemm_xwt<<<256, 256, 0, stream>>>(k16, wk16, bk, (void*)Kh, 0);
    gemm_xwt<<<256, 256, 0, stream>>>(v16, wv16, bv, (void*)Vt, 1);  // V stored transposed

    attn_band<<<512, 256, 0, stream>>>(Qh, Kh, Vt, Ah);

    gemm_xwt<<<256, 256, 0, stream>>>(Ah, wo16, bo, d_out, 2);       // f32 output
}
